// MultiHeadAtt_5566277615815
// MI455X (gfx1250) — matrix operand order repaired and checked by reading, not compiled, not run on hardware
//
#include <hip/hip_runtime.h>
#include <hip/hip_bf16.h>

// ---------------------------------------------------------------------------
// MultiHeadAtt for MI455X (gfx1250): bf16 WMMA projections + flash attention.
// Round 2: native bf16 converts (v_cvt_pk_bf16_f32 path), pipelined loops.
// ---------------------------------------------------------------------------

typedef __bf16 bf16;
typedef __attribute__((ext_vector_type(8)))  __bf16 bf16x8;
typedef __attribute__((ext_vector_type(16))) __bf16 bf16x16;
typedef __attribute__((ext_vector_type(8)))  float  f32x8;
typedef __attribute__((ext_vector_type(4)))  float  f32x4;

constexpr int kBS  = 4;
constexpr int kSEQ = 2048;
constexpr int kDM  = 1024;
constexpr int kNH  = 16;
constexpr int kDK  = 64;

__device__ __forceinline__ bf16 f2bf(float f) {
  return static_cast<bf16>(f);   // native fptrunc -> v_cvt_*_bf16_f32 on gfx1250
}

__device__ __forceinline__ bf16x16 cat16(bf16x8 a, bf16x8 b) {
  return __builtin_shufflevector(a, b, 0,1,2,3,4,5,6,7,8,9,10,11,12,13,14,15);
}

__device__ __forceinline__ f32x8 wmma_bf16(bf16x16 a, bf16x16 b, f32x8 c) {
  // D = A(16x32 bf16) * B(32x16 bf16) + C(16x16 f32)
  return __builtin_amdgcn_wmma_f32_16x16x32_bf16(false, a, false, b,
                                                 (short)0, c, false, false);
}

// ---------------------------------------------------------------------------
// Kernel 1: Wq fp32 -> bf16   (1M elements)
// ---------------------------------------------------------------------------
__global__ void cvt_w_kernel(const float* __restrict__ w, bf16* __restrict__ wbf) {
  int i = blockIdx.x * 256 + threadIdx.x;
  wbf[i] = f2bf(w[i]);
}

// ---------------------------------------------------------------------------
// Kernel 2: projection GEMM  y = X @ Wq^T  (per-wave 16x64 output tile).
//   A-operand: rows of X (fp32 -> bf16 on the fly), contiguous K layout (k = 16*hi + s).
//   B-operand: rows of Wq (bf16), since B^T rows == Wq rows for y = X Wq^T.
//   mat 0 -> qp (scaled by 1/32), mat 1 -> kp, mat 2 -> vT (transposed store).
// ---------------------------------------------------------------------------
__global__ __launch_bounds__(256) void proj_kernel(
    const float* __restrict__ Qf, const float* __restrict__ Kf,
    const float* __restrict__ Vf, const bf16* __restrict__ wbf,
    bf16* __restrict__ qp, bf16* __restrict__ kp, bf16* __restrict__ vT)
{
  const int lane = threadIdx.x & 31;
  const int wave = threadIdx.x >> 5;
  const int l16  = lane & 15;
  const int hi   = lane >> 4;
  const int aoff = hi ? 16 : 0;   // A-layout K split
  const int boff = hi ? 16 : 0;   // B-layout K split

  const int tile = blockIdx.x * 8 + wave;   // 0 .. 24575
  const int mat  = tile >> 13;              // 8192 tiles per matrix
  const int rem  = tile & 8191;
  const int mt   = rem >> 4;                // 512 row tiles (16 rows each)
  const int nt   = rem & 15;                // 16 col tiles (64 cols = 1 head)

  const float* X = (mat == 0) ? Qf : ((mat == 1) ? Kf : Vf);
  const float* xrow = X + (size_t)(mt * 16 + l16) * kDM;

  f32x8 acc[4] = {};
#pragma unroll 2
  for (int kc = 0; kc < kDM; kc += 32) {
    // A: 16 fp32 -> bf16, dims kc + {aoff..aoff+15}
    f32x4 a0 = *(const f32x4*)(xrow + kc + aoff);
    f32x4 a1 = *(const f32x4*)(xrow + kc + aoff + 4);
    f32x4 a2 = *(const f32x4*)(xrow + kc + aoff + 8);
    f32x4 a3 = *(const f32x4*)(xrow + kc + aoff + 12);
    // B tiles for all 4 output column blocks first (independent loads batch
    // into one clause and overlap with the converts below).
    bf16x16 B[4];
#pragma unroll
    for (int j = 0; j < 4; ++j) {
      const bf16* wr = wbf + (size_t)(nt * 64 + j * 16 + l16) * kDM + kc + boff;
      B[j] = cat16(*(const bf16x8*)wr, *(const bf16x8*)(wr + 8));
    }
    bf16x16 A;
#pragma unroll
    for (int i = 0; i < 4; ++i) {
      A[i]      = f2bf(a0[i]);
      A[4 + i]  = f2bf(a1[i]);
      A[8 + i]  = f2bf(a2[i]);
      A[12 + i] = f2bf(a3[i]);
    }
#pragma unroll
    for (int j = 0; j < 4; ++j) acc[j] = wmma_bf16(A, B[j], acc[j]);
  }

  // C layout: VGPR r <-> row (r + 8*hi), col = l16 within 16-wide tile j.
  const int gm0 = mt * 16;
  const int b   = gm0 >> 11;      // 2048 rows per batch; tiles never straddle
  const int s0  = gm0 & 2047;
  const int bh  = b * kNH + nt;   // head == nt (64 cols per head)

  if (mat == 2) {
#pragma unroll
    for (int j = 0; j < 4; ++j) {
      const int d = j * 16 + l16;
      bf16* vp = vT + ((size_t)bh * kDK + d) * kSEQ + s0 + 8 * hi;
      bf16x8 pack;
#pragma unroll
      for (int r = 0; r < 8; ++r) pack[r] = f2bf(acc[j][r]);
      *(bf16x8*)vp = pack;        // one 16B store per tile (16B aligned)
    }
  } else {
    bf16* dst = (mat == 0) ? qp : kp;
    const float sc = (mat == 0) ? 0.03125f : 1.0f;   // fold 1/sqrt(1024) into q
#pragma unroll
    for (int r = 0; r < 8; ++r) {
      const int s = s0 + r + 8 * hi;
      bf16* drow = dst + ((size_t)bh * kSEQ + s) * kDK + l16;
#pragma unroll
      for (int j = 0; j < 4; ++j) drow[j * 16] = f2bf(acc[j][r] * sc);
    }
  }
}

// ---------------------------------------------------------------------------
// Kernel 3: flash attention. One wave = 16 queries of one (b,h); streams keys
// in blocks of 32 with K-operand double buffering and early V loads so global
// latency hides under softmax VALU work. K/V (~33MB bf16) stays in 192MB L2.
// ---------------------------------------------------------------------------
__device__ __forceinline__ void load_k4(const bf16* __restrict__ kbase, int k0,
                                        int l16, int boff, bf16x16 K[4]) {
  const bf16* krA = kbase + (size_t)(k0 + l16) * kDK;        // keys k0..k0+15
  const bf16* krB = kbase + (size_t)(k0 + 16 + l16) * kDK;   // keys +16..+31
  K[0] = cat16(*(const bf16x8*)(krA + boff),      *(const bf16x8*)(krA + boff + 8));
  K[1] = cat16(*(const bf16x8*)(krA + 32 + boff), *(const bf16x8*)(krA + 40 + boff));
  K[2] = cat16(*(const bf16x8*)(krB + boff),      *(const bf16x8*)(krB + boff + 8));
  K[3] = cat16(*(const bf16x8*)(krB + 32 + boff), *(const bf16x8*)(krB + 40 + boff));
}

__global__ __launch_bounds__(256) void attn_kernel(
    const bf16* __restrict__ qp, const bf16* __restrict__ kp,
    const bf16* __restrict__ vT, const int* __restrict__ lens,
    float* __restrict__ out)
{
  const int lane = threadIdx.x & 31;
  const int wave = threadIdx.x >> 5;
  const int l16  = lane & 15;
  const int hi   = lane >> 4;
  const int aoff = hi ? 16 : 0;
  const int boff = hi ? 16 : 0;

  const int blk  = blockIdx.x;     // 1024 blocks
  const int qblk = blk & 15;       // 16 blocks per (b,h), 128 queries each
  const int bh   = blk >> 4;       // 0..63
  const int b    = bh >> 4;
  const int h    = bh & 15;
  const int q0   = qblk * 128 + wave * 16;
  const int len  = lens[b];              // 1..2048 guaranteed
  const int nkb  = (len + 31) >> 5;      // key blocks actually needed

  const bf16* qbase = qp + ((size_t)bh * kSEQ + q0) * kDK;
  const bf16* kbase = kp + (size_t)bh * kSEQ * kDK;
  const bf16* vbase = vT + (size_t)bh * kDK * kSEQ;

  __shared__ __attribute__((aligned(16))) bf16 plds[8][16 * 32];
  bf16* myp = &plds[wave][0];   // per-wave scratch; same-wave DS ops are in-order

  // Q tile (16 x 64), two K=32 A-operands, pre-scaled by 1/32 at projection.
  const bf16* qrow = qbase + (size_t)l16 * kDK;
  bf16x16 qa0 = cat16(*(const bf16x8*)(qrow + aoff),
                      *(const bf16x8*)(qrow + 8 + aoff));
  bf16x16 qa1 = cat16(*(const bf16x8*)(qrow + 32 + aoff),
                      *(const bf16x8*)(qrow + 40 + aoff));

  f32x8 o[4] = {};
  float mrow[8], lrow[8];
#pragma unroll
  for (int r = 0; r < 8; ++r) { mrow[r] = -3.0e38f; lrow[r] = 0.0f; }

  bf16x16 Kcur[4], Knxt[4];
  load_k4(kbase, 0, l16, boff, Kcur);

  for (int kb = 0; kb < nkb; ++kb) {
    const int k0 = kb * 32;

    f32x8 sa = {}, sb = {};
    sa = wmma_bf16(qa0, Kcur[0], sa);
    sa = wmma_bf16(qa1, Kcur[1], sa);
    sb = wmma_bf16(qa0, Kcur[2], sb);
    sb = wmma_bf16(qa1, Kcur[3], sb);

    // Issue next-block K loads and this-block V loads now: their latency
    // hides under the softmax VALU/transcendental work below.
    if (kb + 1 < nkb) load_k4(kbase, k0 + 32, l16, boff, Knxt);
    bf16x16 VB[4];
#pragma unroll
    for (int j = 0; j < 4; ++j) {
      const bf16* vrow = vbase + (size_t)(j * 16 + l16) * kSEQ + k0;
      VB[j] = cat16(*(const bf16x8*)(vrow + boff),
                    *(const bf16x8*)(vrow + boff + 8));
    }

    if (k0 + 32 > len) {           // uniform branch; only last block masks
      const bool mA = (k0 + l16) < len;
      const bool mB = (k0 + 16 + l16) < len;
#pragma unroll
      for (int r = 0; r < 8; ++r) {
        sa[r] = mA ? sa[r] : -1.0e30f;
        sb[r] = mB ? sb[r] : -1.0e30f;
      }
    }

    // online softmax; C-layout row r lives in VGPR r of one 16-lane half,
    // so xor-shuffles over masks 1,2,4,8 reduce per-row across the half.
    float mnew[8], corr[8];
#pragma unroll
    for (int r = 0; r < 8; ++r) {
      float v = fmaxf(sa[r], sb[r]);
      v = fmaxf(v, __shfl_xor(v, 1, 32));
      v = fmaxf(v, __shfl_xor(v, 2, 32));
      v = fmaxf(v, __shfl_xor(v, 4, 32));
      v = fmaxf(v, __shfl_xor(v, 8, 32));
      const float mn = fmaxf(mrow[r], v);
      mnew[r] = mn;
      corr[r] = __expf(mrow[r] - mn);
      mrow[r] = mn;
    }
#pragma unroll
    for (int r = 0; r < 8; ++r) {
      sa[r] = __expf(sa[r] - mnew[r]);
      sb[r] = __expf(sb[r] - mnew[r]);
      float s = sa[r] + sb[r];
      s += __shfl_xor(s, 1, 32);
      s += __shfl_xor(s, 2, 32);
      s += __shfl_xor(s, 4, 32);
      s += __shfl_xor(s, 8, 32);
      lrow[r] = lrow[r] * corr[r] + s;
    }
#pragma unroll
    for (int j = 0; j < 4; ++j)
#pragma unroll
      for (int r = 0; r < 8; ++r) o[j][r] *= corr[r];

    // P: C-layout f32 -> A-layout bf16 via per-wave LDS round trip.
#pragma unroll
    for (int r = 0; r < 8; ++r) {
      const int row = r + 8 * hi;
      myp[row * 32 + l16]      = f2bf(sa[r]);
      myp[row * 32 + 16 + l16] = f2bf(sb[r]);
    }
    bf16x16 PA = cat16(*(const bf16x8*)(myp + l16 * 32 + aoff),
                       *(const bf16x8*)(myp + l16 * 32 + 8 + aoff));

    // O += P (16x32) @ Vblk (32x64); V operands already in registers.
#pragma unroll
    for (int j = 0; j < 4; ++j) o[j] = wmma_bf16(PA, VB[j], o[j]);

#pragma unroll
    for (int i = 0; i < 4; ++i) Kcur[i] = Knxt[i];
  }

#pragma unroll
  for (int r = 0; r < 8; ++r) {
    const float inv = 1.0f / lrow[r];
    const int srow = q0 + r + 8 * hi;
    float* op = out + ((size_t)b * kSEQ + srow) * kDM + h * kDK + l16;
    op[0]  = o[0][r] * inv;
    op[16] = o[1][r] * inv;
    op[32] = o[2][r] * inv;
    op[48] = o[3][r] * inv;
  }
}

// ---------------------------------------------------------------------------
extern "C" void kernel_launch(void* const* d_in, const int* in_sizes, int n_in,
                              void* d_out, int out_size, void* d_ws, size_t ws_size,
                              hipStream_t stream)
{
  const float* Qf  = (const float*)d_in[0];
  const float* Kf  = (const float*)d_in[1];
  const float* Vf  = (const float*)d_in[2];
  const float* Wq  = (const float*)d_in[3];
  const int*   ln  = (const int*)d_in[4];
  float*       out = (float*)d_out;

  // ws layout: [Wq bf16: 2MB][qp:16MB][kp:16MB][vT:16MB]  (~52.4MB total)
  char* ws = (char*)d_ws;
  bf16* wbf = (bf16*)ws;
  bf16* qp  = (bf16*)(ws + (size_t)2 * 1024 * 1024);
  bf16* kp  = qp + (size_t)kBS * kNH * kSEQ * kDK;
  bf16* vT  = kp + (size_t)kBS * kNH * kSEQ * kDK;

  hipLaunchKernelGGL(cvt_w_kernel, dim3(4096), dim3(256), 0, stream, Wq, wbf);
  hipLaunchKernelGGL(proj_kernel,  dim3(3072), dim3(256), 0, stream,
                     Qf, Kf, Vf, wbf, qp, kp, vT);
  hipLaunchKernelGGL(attn_kernel,  dim3(1024), dim3(256), 0, stream,
                     qp, kp, vT, ln, out);
}
